// SRNCell_pad_40535901340154
// MI455X (gfx1250) — compile-verified, matrix operand order checked by reading, not run on hardware
//
#include <hip/hip_runtime.h>
#include <hip/hip_bf16.h>

// Problem constants (match reference)
#define BB 128
#define TT 512
#define II 512
#define HH 2048
#define OO 512
#define STEPS (TT - 2)   // 510
#define KC (II + HH)     // 2560 fused K for [x | h] @ [W_ih | W_hh]^T

typedef __bf16 bf16;
typedef __attribute__((ext_vector_type(16))) __bf16 v16bf;
typedef __attribute__((ext_vector_type(8)))  __bf16 v8bf;
typedef __attribute__((ext_vector_type(8)))  float  v8f;

// Build one WMMA 16-bit operand (v16bf) from the two 8-element K-chunks this
// lane owns per the CDNA5 16-bit A/B layout (lane<16: K {0..7,16..23};
// lane>=16: K {8..15,24..31} -> caller passes the right chunk pointers).
__device__ __forceinline__ v16bf make_ab(const bf16* p0, const bf16* p1) {
    v8bf lo = *reinterpret_cast<const v8bf*>(p0);
    v8bf hi = *reinterpret_cast<const v8bf*>(p1);
    v16bf r;
#pragma unroll
    for (int i = 0; i < 8; ++i) { r[i] = lo[i]; r[i + 8] = hi[i]; }
    return r;
}

// One-time (per launch) prep: bf16 weight conversion, fused bias, pad indices,
// h0 broadcast, Hsel zero-init.
__global__ void srn_prep_kernel(const float* __restrict__ x,
                                const float* __restrict__ h0,
                                const float* __restrict__ W_ih,
                                const float* __restrict__ b_ih,
                                const float* __restrict__ W_hh,
                                const float* __restrict__ b_hh,
                                const float* __restrict__ W_fc,
                                bf16* __restrict__ Wcat,
                                bf16* __restrict__ Wfc,
                                float* __restrict__ bc,
                                int* __restrict__ pad,
                                bf16* __restrict__ Ha,
                                bf16* __restrict__ Hsel) {
    const int tid = blockIdx.x * blockDim.x + threadIdx.x;
    const int nth = gridDim.x * blockDim.x;
    // Fused weight [H, I+H]: cols 0..I-1 = W_ih, cols I.. = W_hh
    for (int idx = tid; idx < HH * KC; idx += nth) {
        const int h = idx / KC;
        const int k = idx - h * KC;
        const float w = (k < II) ? W_ih[h * II + k] : W_hh[h * HH + (k - II)];
        Wcat[idx] = (bf16)w;
    }
    for (int idx = tid; idx < OO * HH; idx += nth) Wfc[idx] = (bf16)W_fc[idx];
    for (int idx = tid; idx < HH; idx += nth) bc[idx] = b_ih[idx] + b_hh[idx];
    for (int idx = tid; idx < BB; idx += nth)
        pad[idx] = (int)x[(size_t)idx * TT * II + (size_t)(TT - 1) * II];
    for (int idx = tid; idx < BB * HH; idx += nth) {
        Ha[idx]   = (bf16)h0[idx % HH];
        Hsel[idx] = (bf16)0.f;
    }
}

// Re-layout x[:, 0:steps, :] (B,T,I) -> Xbf (steps, B, I) in bf16 so each step
// reads a contiguous 128 KiB slab.
__global__ void srn_xconv_kernel(const float* __restrict__ x,
                                 bf16* __restrict__ Xbf) {
    const size_t tid = (size_t)blockIdx.x * blockDim.x + threadIdx.x;
    const size_t nth = (size_t)gridDim.x * blockDim.x;
    const size_t total = (size_t)STEPS * BB * II;
    for (size_t idx = tid; idx < total; idx += nth) {
        const size_t t   = idx / ((size_t)BB * II);
        const size_t rem = idx - t * ((size_t)BB * II);
        const size_t b   = rem / II;
        const size_t i   = rem - b * II;
        Xbf[idx] = (bf16)x[(b * TT + t) * II + i];
    }
}

// One recurrence step: Hnext = tanh([x_t | Hprev] @ Wcat^T + bc), bf16 WMMA
// with fp32 accumulation.
// Grid = H/64 blocks x 8 waves. Block tile 128(M) x 64(N); wave (wm, wn)
// with wm = wave&3, wn = wave>>2 computes the 32x32 subtile at
// rows [32*wm, 32*wm+32) x cols [64*blk + 32*wn, +32): 2x2 WMMA tiles,
// 4 accumulators -> 32 B loaded per WMMA instead of 48 B.
// Also snapshots rows whose pad index equals t into Hsel (replaces the
// reference's masked FC over all timesteps).
__global__ __launch_bounds__(256) void srn_step_kernel(
    const bf16* __restrict__ Xbf, const bf16* __restrict__ Hprev,
    bf16* __restrict__ Hnext, const bf16* __restrict__ Wcat,
    const float* __restrict__ bc, const int* __restrict__ pad,
    bf16* __restrict__ Hsel, int t) {
    const int wave = threadIdx.x >> 5;
    const int lane = threadIdx.x & 31;
    const int half = lane >> 4;
    const int l    = lane & 15;
    const int wm   = wave & 3;                 // M wave coordinate (0..3)
    const int wn   = wave >> 2;                // N wave coordinate (0..1)

    const int rowA0 = wm * 32 + l;             // A rows, M tile 0
    const int rowA1 = wm * 32 + 16 + l;        // A rows, M tile 1
    const int baseN = blockIdx.x * 64 + wn * 32;
    const int col0  = baseN + l;               // B column (output feature), N tile 0
    const int col1  = baseN + 16 + l;          // N tile 1

    const bf16* xrow0 = Xbf + ((size_t)t * BB + rowA0) * II;
    const bf16* xrow1 = Xbf + ((size_t)t * BB + rowA1) * II;
    const bf16* hrow0 = Hprev + (size_t)rowA0 * HH;
    const bf16* hrow1 = Hprev + (size_t)rowA1 * HH;
    const bf16* w0    = Wcat + (size_t)col0 * KC;
    const bf16* w1    = Wcat + (size_t)col1 * KC;

    const int o0 = half * 8;        // this lane's K-chunk offsets in a 32-K block
    const int o1 = 16 + half * 8;

    v8f acc00 = {};   // (Mtile0, Ntile0)
    v8f acc01 = {};   // (Mtile0, Ntile1)
    v8f acc10 = {};   // (Mtile1, Ntile0)
    v8f acc11 = {};   // (Mtile1, Ntile1)

    // K = [0, I): input projection
    for (int kb = 0; kb < II; kb += 32) {
        v16bf a0 = make_ab(xrow0 + kb + o0, xrow0 + kb + o1);
        v16bf a1 = make_ab(xrow1 + kb + o0, xrow1 + kb + o1);
        v16bf b0 = make_ab(w0 + kb + o0, w0 + kb + o1);
        v16bf b1 = make_ab(w1 + kb + o0, w1 + kb + o1);
        acc00 = __builtin_amdgcn_wmma_f32_16x16x32_bf16(false, a0, false, b0,
                                                        (short)0, acc00, false, false);
        acc01 = __builtin_amdgcn_wmma_f32_16x16x32_bf16(false, a0, false, b1,
                                                        (short)0, acc01, false, false);
        acc10 = __builtin_amdgcn_wmma_f32_16x16x32_bf16(false, a1, false, b0,
                                                        (short)0, acc10, false, false);
        acc11 = __builtin_amdgcn_wmma_f32_16x16x32_bf16(false, a1, false, b1,
                                                        (short)0, acc11, false, false);
    }
    // K = [I, I+H): hidden recurrence
    for (int kb = 0; kb < HH; kb += 32) {
        v16bf a0 = make_ab(hrow0 + kb + o0, hrow0 + kb + o1);
        v16bf a1 = make_ab(hrow1 + kb + o0, hrow1 + kb + o1);
        v16bf b0 = make_ab(w0 + II + kb + o0, w0 + II + kb + o1);
        v16bf b1 = make_ab(w1 + II + kb + o0, w1 + II + kb + o1);
        acc00 = __builtin_amdgcn_wmma_f32_16x16x32_bf16(false, a0, false, b0,
                                                        (short)0, acc00, false, false);
        acc01 = __builtin_amdgcn_wmma_f32_16x16x32_bf16(false, a0, false, b1,
                                                        (short)0, acc01, false, false);
        acc10 = __builtin_amdgcn_wmma_f32_16x16x32_bf16(false, a1, false, b0,
                                                        (short)0, acc10, false, false);
        acc11 = __builtin_amdgcn_wmma_f32_16x16x32_bf16(false, a1, false, b1,
                                                        (short)0, acc11, false, false);
    }

    // Epilogue: C/D layout -> tile row = 8*half + r, col = l
    const float bias0 = bc[col0];
    const float bias1 = bc[col1];
#pragma unroll
    for (int r = 0; r < 8; ++r) {
        const int row0 = wm * 32 + 8 * half + r;        // M tile 0
        const int row1 = row0 + 16;                     // M tile 1
        const float v00 = tanhf(acc00[r] + bias0);
        const float v01 = tanhf(acc01[r] + bias1);
        const float v10 = tanhf(acc10[r] + bias0);
        const float v11 = tanhf(acc11[r] + bias1);
        const bf16 b00 = (bf16)v00;
        const bf16 b01 = (bf16)v01;
        const bf16 b10 = (bf16)v10;
        const bf16 b11 = (bf16)v11;
        Hnext[(size_t)row0 * HH + col0] = b00;
        Hnext[(size_t)row0 * HH + col1] = b01;
        Hnext[(size_t)row1 * HH + col0] = b10;
        Hnext[(size_t)row1 * HH + col1] = b11;
        if (pad[row0] == t) {   // snapshot the one timestep the mask keeps
            Hsel[(size_t)row0 * HH + col0] = b00;
            Hsel[(size_t)row0 * HH + col1] = b01;
        }
        if (pad[row1] == t) {
            Hsel[(size_t)row1 * HH + col0] = b10;
            Hsel[(size_t)row1 * HH + col1] = b11;
        }
    }
}

// Final FC over the snapshot: out = Hsel @ W_fc^T + b_fc   (128x512, K=2048)
__global__ __launch_bounds__(256) void srn_fc_kernel(
    const bf16* __restrict__ Hsel, const bf16* __restrict__ Wfc,
    const float* __restrict__ b_fc, float* __restrict__ out) {
    const int wave = threadIdx.x >> 5;
    const int lane = threadIdx.x & 31;
    const int half = lane >> 4;
    const int l    = lane & 15;
    const int rowA  = wave * 16 + l;
    const int baseN = blockIdx.x * 32;
    const int col0  = baseN + l;
    const int col1  = baseN + 16 + l;

    const bf16* hrow = Hsel + (size_t)rowA * HH;
    const bf16* w0   = Wfc + (size_t)col0 * HH;
    const bf16* w1   = Wfc + (size_t)col1 * HH;

    const int o0 = half * 8;
    const int o1 = 16 + half * 8;

    v8f acc0 = {};
    v8f acc1 = {};
    for (int kb = 0; kb < HH; kb += 32) {
        v16bf a  = make_ab(hrow + kb + o0, hrow + kb + o1);
        v16bf b0 = make_ab(w0 + kb + o0, w0 + kb + o1);
        v16bf b1 = make_ab(w1 + kb + o0, w1 + kb + o1);
        acc0 = __builtin_amdgcn_wmma_f32_16x16x32_bf16(false, a, false, b0,
                                                       (short)0, acc0, false, false);
        acc1 = __builtin_amdgcn_wmma_f32_16x16x32_bf16(false, a, false, b1,
                                                       (short)0, acc1, false, false);
    }
    const float bias0 = b_fc[col0];
    const float bias1 = b_fc[col1];
#pragma unroll
    for (int r = 0; r < 8; ++r) {
        const int row = wave * 16 + 8 * half + r;
        out[(size_t)row * OO + col0] = acc0[r] + bias0;
        out[(size_t)row * OO + col1] = acc1[r] + bias1;
    }
}

extern "C" void kernel_launch(void* const* d_in, const int* in_sizes, int n_in,
                              void* d_out, int out_size, void* d_ws, size_t ws_size,
                              hipStream_t stream) {
    (void)in_sizes; (void)n_in; (void)out_size; (void)ws_size;
    const float* x    = (const float*)d_in[0];
    const float* h0   = (const float*)d_in[1];
    const float* W_ih = (const float*)d_in[2];
    const float* b_ih = (const float*)d_in[3];
    const float* W_hh = (const float*)d_in[4];
    const float* b_hh = (const float*)d_in[5];
    const float* W_fc = (const float*)d_in[6];
    const float* b_fc = (const float*)d_in[7];
    float* out = (float*)d_out;

    // Workspace carve-up (256B aligned), ~81 MB total
    char* ws = (char*)d_ws;
    size_t off = 0;
    auto alloc = [&](size_t bytes) -> void* {
        void* p = ws + off;
        off = (off + bytes + 255) & ~(size_t)255;
        return p;
    };
    bf16*  Wcat = (bf16*) alloc((size_t)HH * KC * sizeof(bf16));
    bf16*  Wfc  = (bf16*) alloc((size_t)OO * HH * sizeof(bf16));
    float* bc   = (float*)alloc((size_t)HH * sizeof(float));
    int*   pad  = (int*)  alloc((size_t)BB * sizeof(int));
    bf16*  Xbf  = (bf16*) alloc((size_t)STEPS * BB * II * sizeof(bf16));
    bf16*  Ha   = (bf16*) alloc((size_t)BB * HH * sizeof(bf16));
    bf16*  Hb   = (bf16*) alloc((size_t)BB * HH * sizeof(bf16));
    bf16*  Hsel = (bf16*) alloc((size_t)BB * HH * sizeof(bf16));

    srn_prep_kernel<<<512, 256, 0, stream>>>(x, h0, W_ih, b_ih, W_hh, b_hh,
                                             W_fc, Wcat, Wfc, bc, pad, Ha, Hsel);
    srn_xconv_kernel<<<1024, 256, 0, stream>>>(x, Xbf);

    bf16* hp = Ha;
    bf16* hn = Hb;
    for (int t = 0; t < STEPS; ++t) {
        srn_step_kernel<<<HH / 64, 256, 0, stream>>>(Xbf, hp, hn, Wcat, bc,
                                                     pad, Hsel, t);
        bf16* tmp = hp; hp = hn; hn = tmp;
    }
    srn_fc_kernel<<<OO / 32, 256, 0, stream>>>(Hsel, Wfc, b_fc, out);
}
